// CausalSelfAttention_81638738362535
// MI455X (gfx1250) — matrix operand order repaired and checked by reading, not compiled, not run on hardware
//
#include <hip/hip_runtime.h>

#define B_    2
#define T_    2048
#define EMB_  1024
#define H_    16
#define D_    64
#define M_TOT (B_ * T_)     /* 4096 */
#define N_QKV (3 * H_ * D_) /* 3072 */
#define HD_   (H_ * D_)     /* 1024 */

#define BROW  40            /* padded LDS row stride (elems): 16B aligned, conflict-free */
#define PANEL_BYTES (64 * BROW * 2)   /* 5120 B per staged B panel */

typedef __bf16 bf16;
typedef __attribute__((ext_vector_type(16))) __bf16 v16bf;
typedef __attribute__((ext_vector_type(8)))  __bf16 bf16x8;
typedef __attribute__((ext_vector_type(8)))  float  v8f;

typedef unsigned int u32x4 __attribute__((ext_vector_type(4)));
typedef int          i32x4 __attribute__((ext_vector_type(4)));
typedef int          i32x8 __attribute__((ext_vector_type(8)));

union AccU { v8f v; float f[8]; };

#if __has_builtin(__builtin_amdgcn_tensor_load_to_lds)
#define USE_TDM 1
#else
#define USE_TDM 0
#endif

// The TDM engine writes LDS behind the compiler's back. Escape the buffer pointer
// into an asm memory barrier so "no stores -> loads are undef" reasoning is blocked.
#define TDM_MEM_CLOBBER(p) asm volatile("" : : "r"((const void*)(p)) : "memory")

// A-operand (16x32, M x K): lane holds row m = lane%16.
__device__ __forceinline__ v16bf load_a16(const bf16* row, int kbase, int lane) {
  int kb = kbase + ((lane >> 4) << 4);
  union { v16bf v; bf16x8 h[2]; } u;
  u.h[0] = *(const bf16x8*)(row + kb);
  u.h[1] = *(const bf16x8*)(row + kb + 8);
  return u.v;
}

// B-operand (32x16, K x N): lane holds column n = lane%16; 16 contiguous K values.
__device__ __forceinline__ v16bf load_b16(const bf16* row, int kbase, int lane) {
  int kb = kbase + ((lane >> 4) << 4);
  union { v16bf v; bf16x8 h[2]; } u;
  u.h[0] = *(const bf16x8*)(row + kb);
  u.h[1] = *(const bf16x8*)(row + kb + 8);
  return u.v;
}

__device__ __forceinline__ v8f wmma_bf16(v16bf a, v16bf b, v8f c) {
  return __builtin_amdgcn_wmma_f32_16x16x32_bf16(false, a, false, b, (short)0, c,
                                                 false, false);
}

#if USE_TDM
// TDM: DMA a 64-row x 32-elem bf16 tile (row stride = 1024 elems in memory) into LDS
// at byte offset ldsOff, hardware-padding each 64B row by 16B -> 80B LDS row stride.
__device__ __forceinline__ void tdm_load_panel(const void* gsrc, unsigned ldsOff,
                                               unsigned nrows) {
  unsigned long long ga = (unsigned long long)(size_t)gsrc;
  union { u32x4 v; unsigned e[4]; } g0 = {};
  g0.e[0] = 1u;                                   // count = 1 (valid descriptor)
  g0.e[1] = ldsOff;                               // lds_addr (bytes)
  g0.e[2] = (unsigned)ga;                         // global_addr[31:0]
  g0.e[3] = (unsigned)((ga >> 32) & 0x01ffffffu)  // global_addr[56:32]
          | (2u << 30);                           // type = 2 ("image")
  union { i32x8 v; unsigned e[8]; } g1 = {};
  g1.e[0] = (1u << 16)        // data_size = 2 bytes
          | (1u << 20)        // pad_enable
          | (3u << 22)        // pad_interval: 16 DWORDs (one 32-elem row)
          | (3u << 25);       // pad_amount: 4 DWORDs (16B)
  g1.e[1] = (EMB_ & 0xffffu) << 16;               // tensor_dim0 = 1024 (bits 63:48)
  g1.e[2] = (nrows & 0xffffu) << 16;              // tensor_dim1 (bits 111:80)
  g1.e[3] = 32u << 16;                            // tile_dim0 = 32 (bits 127:112)
  g1.e[4] = 64u;                                  // tile_dim1 = 64 rows
  g1.e[5] = EMB_;                                 // tensor_dim0_stride = 1024
  i32x4 z4 = {};
#if __clang_major__ >= 23
  i32x8 z8 = {};
  __builtin_amdgcn_tensor_load_to_lds(g0.v, g1.v, z4, z4, z8, 0);
#else
  __builtin_amdgcn_tensor_load_to_lds(g0.v, g1.v, z4, z4, 0);
#endif
}
#endif

// ---------------- prep kernels ----------------

__global__ void k_conv_x(const float* __restrict__ x, bf16* __restrict__ xb, int n) {
  int i = blockIdx.x * blockDim.x + threadIdx.x;
  if (i < n) xb[i] = (bf16)x[i];
}

// wt[n*K + k] = (bf16) w[k*N + n]
__global__ void k_transpose_w(const float* __restrict__ w, bf16* __restrict__ wt,
                              int K, int N) {
  int i = blockIdx.x * blockDim.x + threadIdx.x;
  if (i < K * N) {
    int n = i / K;
    int k = i - n * K;
    wt[i] = (bf16)w[(size_t)k * N + n];
  }
}

// ---- QKV GEMM: block = 256 rows x 64 cols; B panel TDM/LDS staged, double-buffered ----

__global__ void __launch_bounds__(256, 1)
k_gemm_qkv(const bf16* __restrict__ xb, const bf16* __restrict__ w1t,
           const float* __restrict__ bqkv,
           bf16* __restrict__ qb, bf16* __restrict__ kbuf, bf16* __restrict__ vt) {
  __shared__ __align__(16) bf16 bsh[2][64 * BROW];
  int lane = threadIdx.x & 31, wave = threadIdx.x >> 5;
  int tn  = blockIdx.x % (N_QKV / 64);          // 48 column panels
  int tmg = blockIdx.x / (N_QKV / 64);          // 16 row groups
  int tm  = tmg * 8 + wave;                     // 32-row strip per wave

  const bf16* ar0 = xb + (size_t)(tm * 32 + (lane & 15)) * EMB_;
  const bf16* ar1 = ar0 + (size_t)16 * EMB_;
  const bf16* bpanel = w1t + (size_t)(tn * 64) * EMB_;

#if USE_TDM
  if (threadIdx.x == 0) {
    tdm_load_panel(bpanel, 0u, N_QKV);
    __builtin_amdgcn_s_wait_tensorcnt(0);
  }
  __syncthreads();
  TDM_MEM_CLOBBER(&bsh[0][0]);
#else
  int col = threadIdx.x >> 2;
  int kp  = (threadIdx.x & 3) * 8;
  const bf16* bsrc = bpanel + (size_t)col * EMB_ + kp;
  *(bf16x8*)&bsh[0][col * BROW + kp] = *(const bf16x8*)bsrc;
  __syncthreads();
#endif

  AccU acc[2][4] = {};
  int cur = 0;
#pragma unroll 1
  for (int k = 0; k < EMB_; k += 32) {
    bool more = (k + 32) < EMB_;
#if USE_TDM
    if (more && threadIdx.x == 0)
      tdm_load_panel(bpanel + k + 32, (unsigned)((cur ^ 1) * PANEL_BYTES), N_QKV);
#else
    bf16x8 bnext = {};
    if (more) bnext = *(const bf16x8*)(bsrc + k + 32);
#endif
    __builtin_prefetch((const void*)(ar0 + k + 128), 0, 1);
    __builtin_prefetch((const void*)(ar1 + k + 128), 0, 1);

    v16bf a0 = load_a16(ar0, k, lane);
    v16bf a1 = load_a16(ar1, k, lane);
    v16bf bt[4];
#pragma unroll
    for (int c = 0; c < 4; c++)
      bt[c] = load_b16(&bsh[cur][(c * 16 + (lane & 15)) * BROW], 0, lane);
#pragma unroll
    for (int c = 0; c < 4; c++) {
      acc[0][c].v = wmma_bf16(a0, bt[c], acc[0][c].v);
      acc[1][c].v = wmma_bf16(a1, bt[c], acc[1][c].v);
    }
    if (more) {
#if USE_TDM
      if (threadIdx.x == 0) __builtin_amdgcn_s_wait_tensorcnt(0);
      __syncthreads();
      TDM_MEM_CLOBBER(&bsh[0][0]);
#else
      *(bf16x8*)&bsh[cur ^ 1][col * BROW + kp] = bnext;
      __syncthreads();
#endif
    }
    cur ^= 1;
  }

#pragma unroll
  for (int c = 0; c < 4; c++) {
    int n = tn * 64 + c * 16 + (lane & 15);
    float bias = bqkv[n];
    int sel = n >> 10;            // 0:q 1:k 2:v
    int within = n & 1023;
    int h = within >> 6, d = within & 63;
    bf16* base = (sel == 0) ? qb : ((sel == 1) ? kbuf : vt);
#pragma unroll
    for (int r = 0; r < 2; r++) {
#pragma unroll
      for (int i = 0; i < 8; i++) {
        int m = tm * 32 + r * 16 + i + ((lane >> 4) << 3);
        int b = m >> 11, t = m & 2047;
        size_t bh = (size_t)(b * H_ + h);
        size_t idx = (sel == 2) ? ((bh * D_ + d) * T_ + t)    // V stored transposed
                                : ((bh * T_ + t) * D_ + d);
        base[idx] = (bf16)(acc[r][c].f[i] + bias);
      }
    }
  }
}

// ---------------- flash attention: one (b,h,16-query-row tile) per wave ----------------

__global__ void __launch_bounds__(256, 1)
k_attn(const bf16* __restrict__ qb, const bf16* __restrict__ kbuf,
       const bf16* __restrict__ vt, bf16* __restrict__ ao) {
  __shared__ __align__(16) bf16 psh[8][16 * 32];   // per-wave P staging (C->A relayout)
  int lane = threadIdx.x & 31, wave = threadIdx.x >> 5;
  int job = blockIdx.x * 8 + wave;                 // 4096 jobs
  int qt = job & 127;                              // query tile (of 16 rows)
  int bh = job >> 7;                               // b*H + h
  const bf16* Q = qb   + ((size_t)bh * T_ + qt * 16) * D_;
  const bf16* K = kbuf + (size_t)bh * T_ * D_;
  const bf16* V = vt   + (size_t)bh * D_ * T_;     // [64][2048]

  const bf16* qrow = Q + (size_t)(lane & 15) * D_;
  v16bf qa0 = load_a16(qrow, 0, lane);             // d = 0..31
  v16bf qa1 = load_a16(qrow, 32, lane);            // d = 32..63

  AccU o0 = {}, o1 = {}, o2 = {}, o3 = {};
  float mrow[8], lrow[8];
#pragma unroll
  for (int i = 0; i < 8; i++) { mrow[i] = -1e30f; lrow[i] = 0.0f; }

  int nblk = ((qt * 16 + 15) >> 5) + 1;            // causal: key blocks of 32
#pragma unroll 1
  for (int j = 0; j < nblk; j++) {
    int k0 = j * 32;
    const bf16* kr0 = K + (size_t)(k0 + (lane & 15)) * D_;
    const bf16* kr1 = K + (size_t)(k0 + 16 + (lane & 15)) * D_;
    AccU s0 = {}, s1 = {};
    s0.v = wmma_bf16(qa0, load_b16(kr0, 0, lane), s0.v);
    s0.v = wmma_bf16(qa1, load_b16(kr0, 32, lane), s0.v);
    s1.v = wmma_bf16(qa0, load_b16(kr1, 0, lane), s1.v);
    s1.v = wmma_bf16(qa1, load_b16(kr1, 32, lane), s1.v);

    int key0 = k0 + (lane & 15);
    int key1 = key0 + 16;
#pragma unroll
    for (int i = 0; i < 8; i++) {
      int qg = qt * 16 + i + ((lane >> 4) << 3);
      float a0 = (key0 <= qg) ? s0.f[i] * 0.125f : -1e30f;   // SCALE = 1/sqrt(64)
      float a1 = (key1 <= qg) ? s1.f[i] * 0.125f : -1e30f;
      float mx = fmaxf(a0, a1);
#pragma unroll
      for (int off = 1; off < 16; off <<= 1)
        mx = fmaxf(mx, __shfl_xor(mx, off, 32));
      float mn = fmaxf(mrow[i], mx);
      float alpha = __expf(mrow[i] - mn);
      mrow[i] = mn;
      float p0 = __expf(a0 - mn);
      float p1 = __expf(a1 - mn);
      float rs = p0 + p1;
#pragma unroll
      for (int off = 1; off < 16; off <<= 1) rs += __shfl_xor(rs, off, 32);
      lrow[i] = lrow[i] * alpha + rs;
      o0.f[i] *= alpha; o1.f[i] *= alpha; o2.f[i] *= alpha; o3.f[i] *= alpha;
      int prow = i + ((lane >> 4) << 3);
      psh[wave][prow * 32 + (lane & 15)]      = (bf16)p0;
      psh[wave][prow * 32 + 16 + (lane & 15)] = (bf16)p1;
    }
    v16bf pa = load_a16(&psh[wave][(lane & 15) * 32], 0, lane);
    o0.v = wmma_bf16(pa, load_b16(V + (size_t)(0 * 16 + (lane & 15)) * T_, k0, lane), o0.v);
    o1.v = wmma_bf16(pa, load_b16(V + (size_t)(1 * 16 + (lane & 15)) * T_, k0, lane), o1.v);
    o2.v = wmma_bf16(pa, load_b16(V + (size_t)(2 * 16 + (lane & 15)) * T_, k0, lane), o2.v);
    o3.v = wmma_bf16(pa, load_b16(V + (size_t)(3 * 16 + (lane & 15)) * T_, k0, lane), o3.v);
  }

  int b = bh >> 4, h = bh & 15;
#pragma unroll
  for (int i = 0; i < 8; i++) {
    float inv = 1.0f / lrow[i];
    int t = qt * 16 + i + ((lane >> 4) << 3);
    size_t rowoff = ((size_t)(b * T_ + t)) * HD_ + h * D_;
    ao[rowoff + 0 * 16 + (lane & 15)] = (bf16)(o0.f[i] * inv);
    ao[rowoff + 1 * 16 + (lane & 15)] = (bf16)(o1.f[i] * inv);
    ao[rowoff + 2 * 16 + (lane & 15)] = (bf16)(o2.f[i] * inv);
    ao[rowoff + 3 * 16 + (lane & 15)] = (bf16)(o3.f[i] * inv);
  }
}

// ---- out projection: block = 256 rows x 64 cols; B panel TDM/LDS staged -> fp32 ----

__global__ void __launch_bounds__(256, 1)
k_gemm_out(const bf16* __restrict__ ao, const bf16* __restrict__ w2t,
           const float* __restrict__ bout, float* __restrict__ out) {
  __shared__ __align__(16) bf16 bsh[2][64 * BROW];
  int lane = threadIdx.x & 31, wave = threadIdx.x >> 5;
  int tn  = blockIdx.x & 15;                    // 16 column panels
  int tmg = blockIdx.x >> 4;                    // 16 row groups
  int tm  = tmg * 8 + wave;

  const bf16* ar0 = ao + (size_t)(tm * 32 + (lane & 15)) * HD_;
  const bf16* ar1 = ar0 + (size_t)16 * HD_;
  const bf16* bpanel = w2t + (size_t)(tn * 64) * HD_;

#if USE_TDM
  if (threadIdx.x == 0) {
    tdm_load_panel(bpanel, 0u, HD_);
    __builtin_amdgcn_s_wait_tensorcnt(0);
  }
  __syncthreads();
  TDM_MEM_CLOBBER(&bsh[0][0]);
#else
  int col = threadIdx.x >> 2;
  int kp  = (threadIdx.x & 3) * 8;
  const bf16* bsrc = bpanel + (size_t)col * HD_ + kp;
  *(bf16x8*)&bsh[0][col * BROW + kp] = *(const bf16x8*)bsrc;
  __syncthreads();
#endif

  AccU acc[2][4] = {};
  int cur = 0;
#pragma unroll 1
  for (int k = 0; k < HD_; k += 32) {
    bool more = (k + 32) < HD_;
#if USE_TDM
    if (more && threadIdx.x == 0)
      tdm_load_panel(bpanel + k + 32, (unsigned)((cur ^ 1) * PANEL_BYTES), HD_);
#else
    bf16x8 bnext = {};
    if (more) bnext = *(const bf16x8*)(bsrc + k + 32);
#endif
    __builtin_prefetch((const void*)(ar0 + k + 128), 0, 1);
    __builtin_prefetch((const void*)(ar1 + k + 128), 0, 1);

    v16bf a0 = load_a16(ar0, k, lane);
    v16bf a1 = load_a16(ar1, k, lane);
    v16bf bt[4];
#pragma unroll
    for (int c = 0; c < 4; c++)
      bt[c] = load_b16(&bsh[cur][(c * 16 + (lane & 15)) * BROW], 0, lane);
#pragma unroll
    for (int c = 0; c < 4; c++) {
      acc[0][c].v = wmma_bf16(a0, bt[c], acc[0][c].v);
      acc[1][c].v = wmma_bf16(a1, bt[c], acc[1][c].v);
    }
    if (more) {
#if USE_TDM
      if (threadIdx.x == 0) __builtin_amdgcn_s_wait_tensorcnt(0);
      __syncthreads();
      TDM_MEM_CLOBBER(&bsh[0][0]);
#else
      *(bf16x8*)&bsh[cur ^ 1][col * BROW + kp] = bnext;
      __syncthreads();
#endif
    }
    cur ^= 1;
  }

#pragma unroll
  for (int c = 0; c < 4; c++) {
    int n = tn * 64 + c * 16 + (lane & 15);
    float bias = bout[n];
#pragma unroll
    for (int r = 0; r < 2; r++) {
#pragma unroll
      for (int i = 0; i < 8; i++) {
        int m = tm * 32 + r * 16 + i + ((lane >> 4) << 3);
        out[(size_t)m * HD_ + n] = acc[r][c].f[i] + bias;
      }
    }
  }
}

// ---------------- launch ----------------

extern "C" void kernel_launch(void* const* d_in, const int* in_sizes, int n_in,
                              void* d_out, int out_size, void* d_ws, size_t ws_size,
                              hipStream_t stream) {
  const float* x    = (const float*)d_in[0];   // [2,2048,1024]
  const float* Wqkv = (const float*)d_in[1];   // [1024,3072]
  const float* bqkv = (const float*)d_in[2];   // [3072]
  const float* Wout = (const float*)d_in[3];   // [1024,1024]
  const float* bout = (const float*)d_in[4];   // [1024]
  float* out = (float*)d_out;                  // [2,2048,1024] fp32

  bf16* xb  = (bf16*)d_ws;                               // [4096][1024]
  bf16* w1t = xb  + (size_t)M_TOT * EMB_;                // [3072][1024]
  bf16* w2t = w1t + (size_t)N_QKV * EMB_;                // [1024][1024]
  bf16* qb  = w2t + (size_t)HD_ * EMB_;                  // [B,H,T,64]
  bf16* kb  = qb  + (size_t)B_ * H_ * T_ * D_;           // [B,H,T,64]
  bf16* vt  = kb  + (size_t)B_ * H_ * T_ * D_;           // [B,H,64,T]
  bf16* ao  = vt  + (size_t)B_ * H_ * T_ * D_;           // [4096][1024]

  k_conv_x<<<(M_TOT * EMB_) / 256, 256, 0, stream>>>(x, xb, M_TOT * EMB_);
  k_transpose_w<<<(EMB_ * N_QKV) / 256, 256, 0, stream>>>(Wqkv, w1t, EMB_, N_QKV);
  k_transpose_w<<<(EMB_ * HD_) / 256, 256, 0, stream>>>(Wout, w2t, EMB_, HD_);

  k_gemm_qkv<<<(N_QKV / 64) * (M_TOT / 32 / 8), 256, 0, stream>>>(xb, w1t, bqkv,
                                                                  qb, kb, vt);
  k_attn<<<(B_ * H_ * (T_ / 16)) / 8, 256, 0, stream>>>(qb, kb, vt, ao);
  k_gemm_out<<<(HD_ / 64) * (M_TOT / 32 / 8), 256, 0, stream>>>(ao, w2t, bout, out);
}
